// RelGraphConv_47373489275161
// MI455X (gfx1250) — compile-verified, matrix operand order checked by reading, not run on hardware
//
#include <hip/hip_runtime.h>
#include <hip/hip_bf16.h>
#include <stdint.h>

// ---------------------------------------------------------------------------
// RGCN basis layer for MI455X (gfx1250, wave32).
//   Phase A: edge scatter with hardware f32 global atomics (memory bound).
//   Phase B: fused GEMM [N,2304] @ [2304,256] via v_wmma_f32_16x16x32_bf16,
//            f32 accumulation, double-buffered LDS A staging (1 barrier per
//            32-K step), software prefetch, bias + ReLU epilogue.
// ---------------------------------------------------------------------------

typedef __attribute__((ext_vector_type(16))) __bf16 v16bf;
typedef __attribute__((ext_vector_type(2)))  __bf16 v2bf;
typedef __attribute__((ext_vector_type(8)))  float  v8f;
typedef __attribute__((ext_vector_type(4)))  float  f32x4;

#define DFEAT   256       // IN_FEAT == OUT_FEAT
#define NBASES  8
#define KTOTAL  (NBASES * DFEAT + DFEAT)   // 2304 stacked K (8 bases + loop)

static __device__ inline unsigned short f2bf(float f) {
  __bf16 h = (__bf16)f;                 // RNE f32 -> bf16
  unsigned short u;
  __builtin_memcpy(&u, &h, 2);
  return u;
}

// Pack two f32 into packed bf16 pair (lowers to v_cvt_pk_bf16_f32).
static __device__ inline unsigned pk2bf(float x, float y) {
  v2bf v = {(__bf16)x, (__bf16)y};
  unsigned u;
  __builtin_memcpy(&u, &v, 4);
  return u;
}

// K reorder inside a 32-wide k-tile so that each half-wave reads 16
// contiguous bf16: pos order = [K0..7, K16..23, K8..15, K24..31]
// (matches the 16-bit WMMA A/B VGPR striping: lanes 0-15 hold K{0..7,16..23},
//  lanes 16-31 hold K{8..15,24..31}).
static __device__ inline int kposmap(int kk) {
  int q = (kk >> 3) & 3;
  return (q == 1 || q == 2) ? (kk ^ 24) : kk;
}

// --- Kernel 1: pack stacked weights [2304][256] f32 -> bf16 fragment layout.
// wpack[(g*256 + n)*32 + pos] where g = k/32, pos = kposmap(k%32).
__global__ void rgcn_pack_w(const float* __restrict__ weight,      // [8,256,256]
                            const float* __restrict__ loopw,       // [256,256]
                            unsigned short* __restrict__ wpack) {
  int idx = blockIdx.x * 256 + threadIdx.x;       // k*256 + n, 2304*256 total
  int k = idx >> 8;
  int n = idx & 255;
  float v = (k < NBASES * DFEAT) ? weight[(size_t)k * DFEAT + n]
                                 : loopw[(size_t)(k - NBASES * DFEAT) * DFEAT + n];
  int g   = k >> 5;
  int pos = kposmap(k & 31);
  wpack[((size_t)g * DFEAT + n) * 32 + pos] = f2bf(v);
}

// --- Kernel 2: zero the agg workspace (float4-vectorized grid stride).
__global__ void rgcn_zero(float4* __restrict__ p, size_t n4) {
  size_t i = (size_t)blockIdx.x * blockDim.x + threadIdx.x;
  size_t stride = (size_t)gridDim.x * blockDim.x;
  float4 z = make_float4(0.f, 0.f, 0.f, 0.f);
  for (; i < n4; i += stride) p[i] = z;
}

// --- Kernel 3: edge scatter. One block (256 threads) per edge; thread d owns
// feature d; nb hardware f32 atomics into agg[b][dst][d].
__global__ __launch_bounds__(256) void rgcn_scatter(
    const float* __restrict__ x, const int* __restrict__ src,
    const int* __restrict__ dst, const int* __restrict__ et,
    const float* __restrict__ norm, const float* __restrict__ wcomp,
    float* __restrict__ agg, int b0, int nb, int n_nodes, int n_edges) {
  int e = blockIdx.x;
  if (e >= n_edges) return;
  __shared__ float sc[NBASES];
  __shared__ int ss, sd;
  __shared__ float sn;
  if (threadIdx.x == 0) { ss = src[e]; sd = dst[e]; sn = norm[e]; }
  if (threadIdx.x < (unsigned)nb) {
    int t = et[e];
    sc[threadIdx.x] = wcomp[t * NBASES + b0 + threadIdx.x];
  }
  __syncthreads();
  int f = threadIdx.x;
  float xs = x[(size_t)ss * DFEAT + f] * sn;
  size_t base = (size_t)sd * DFEAT + f;
  size_t per = (size_t)n_nodes * DFEAT;
  for (int b = 0; b < nb; ++b) {
    __hip_atomic_fetch_add(agg + (size_t)b * per + base, xs * sc[b],
                           __ATOMIC_RELAXED, __HIP_MEMORY_SCOPE_AGENT);
  }
}

// --- Kernel 4: fused GEMM. Block = 256 threads = 8 waves; block tile is
// 32 rows x 256 cols; wave w owns columns [32w, 32w+32) as 2x2 WMMA tiles.
// A (f32, streamed once -> non-temporal) is staged through a double-buffered
// LDS tile with inline bf16 conversion. The t-loop (8 k-tiles per 256-K
// chunk) is fully unrolled so A/B addresses fold into immediate offsets;
// only the t==7 step selects the next chunk's base pointer.
__global__ __launch_bounds__(256) void rgcn_gemm(
    const float* __restrict__ agg, const float* __restrict__ x,
    const unsigned short* __restrict__ wpack, const float* __restrict__ bias,
    float* __restrict__ out, int n_nodes, int ccb, int nb,
    int include_x, int first, int last) {
  __shared__ __align__(16) unsigned short ldsA[2][32 * 32];  // [buf][row*32+pos]

  const int tid  = threadIdx.x;
  const int wave = tid >> 5;
  const int lane = tid & 31;
  const int hi   = (lane >> 4) & 1;   // half-wave select
  const int l15  = lane & 15;
  const int m0   = blockIdx.x * 32;

  // A-tile loader role: thread loads 4 consecutive f32 of one row.
  const int wrow = tid >> 3;          // 0..31
  const int wkq  = (tid & 7) * 4;     // k quad within 32
  const int wpos = kposmap(wkq);      // contiguous quad maps contiguously
  int arow = m0 + wrow;
  if (arow >= n_nodes) arow = n_nodes - 1;   // clamp; result discarded
  const size_t aoff = (size_t)arow * DFEAT + wkq;  // + t*32 per k-tile

  v8f acc[2][2];
  #pragma unroll
  for (int f = 0; f < 2; ++f)
    #pragma unroll
    for (int j = 0; j < 2; ++j)
      acc[f][j] = (v8f){0.f, 0.f, 0.f, 0.f, 0.f, 0.f, 0.f, 0.f};

  const int nchunks = nb + (include_x ? 1 : 0);
  const size_t chunkStride = (size_t)n_nodes * DFEAT;

  // stage tile (cc=0, t=0) into buf0
  {
    const float* A0 = (0 < nb) ? agg : x;
    f32x4 a4 = __builtin_nontemporal_load((const f32x4*)(A0 + aoff));
    uint2 p;
    p.x = pk2bf(a4[0], a4[1]);
    p.y = pk2bf(a4[2], a4[3]);
    *(uint2*)&ldsA[0][wrow * 32 + wpos] = p;
  }

  for (int cc = 0; cc < nchunks; ++cc) {
    const float* Asrc = (cc < nb) ? agg + (size_t)cc * chunkStride : x;
    const bool moreChunks = (cc + 1 < nchunks);
    const float* AsrcN = (cc + 1 < nb) ? agg + (size_t)(cc + 1) * chunkStride : x;
    const int gbase = ((cc < nb) ? (ccb + cc) * 8 : NBASES * 8);
    const unsigned short* wb =
        wpack + ((size_t)gbase * DFEAT + wave * 32 + l15) * 32 + hi * 16;

    #pragma unroll
    for (int t = 0; t < 8; ++t) {
      __syncthreads();                  // tile (cc,t) visible in buf (t&1)

      // prefetch next A tile (constant-offset load except at chunk tail)
      f32x4 nxt;
      bool more = true;
      if (t < 7) {
        nxt = __builtin_nontemporal_load(
            (const f32x4*)(Asrc + aoff + (t + 1) * 32));
      } else if (moreChunks) {
        nxt = __builtin_nontemporal_load((const f32x4*)(AsrcN + aoff));
      } else {
        more = false;
      }

      // A fragments: lane holds row (f*16 + l15), half-wave picks K subset.
      v16bf afr[2];
      #pragma unroll
      for (int f = 0; f < 2; ++f) {
        const uint4* pa =
            (const uint4*)&ldsA[t & 1][(f * 16 + l15) * 32 + hi * 16];
        union { v16bf v; uint4 u[2]; } ua;
        ua.u[0] = pa[0];
        ua.u[1] = pa[1];
        afr[f] = ua.v;
      }
      // B fragments from packed weights (hot in L2/WGP$); immediate offsets.
      v16bf bfr[2];
      #pragma unroll
      for (int j = 0; j < 2; ++j) {
        const uint4* pb =
            (const uint4*)(wb + ((size_t)t * DFEAT + j * 16) * 32);
        union { v16bf v; uint4 u[2]; } ub;
        ub.u[0] = pb[0];
        ub.u[1] = pb[1];
        bfr[j] = ub.v;
      }

      // stage next tile into the other buffer (guarded by next syncthreads)
      if (more) {
        uint2 p;
        p.x = pk2bf(nxt[0], nxt[1]);
        p.y = pk2bf(nxt[2], nxt[3]);
        *(uint2*)&ldsA[(t + 1) & 1][wrow * 32 + wpos] = p;
      }

      #pragma unroll
      for (int f = 0; f < 2; ++f)
        #pragma unroll
        for (int j = 0; j < 2; ++j)
          acc[f][j] = __builtin_amdgcn_wmma_f32_16x16x32_bf16(
              false, afr[f], false, bfr[j], (short)0, acc[f][j], false, false);
    }
  }

  // Epilogue: C layout = VGPR r -> row (r + hi*8), lane -> col l15.
  #pragma unroll
  for (int f = 0; f < 2; ++f) {
    #pragma unroll
    for (int j = 0; j < 2; ++j) {
      int col = wave * 32 + j * 16 + l15;
      #pragma unroll
      for (int r = 0; r < 8; ++r) {
        int row = m0 + f * 16 + r + hi * 8;
        if (row < n_nodes) {
          size_t o = (size_t)row * DFEAT + col;
          float v = acc[f][j][r];
          if (!first) v += out[o];
          if (last) {
            v += bias[col];
            v = v > 0.f ? v : 0.f;
          }
          out[o] = v;
        }
      }
    }
  }
}

extern "C" void kernel_launch(void* const* d_in, const int* in_sizes, int n_in,
                              void* d_out, int out_size, void* d_ws,
                              size_t ws_size, hipStream_t stream) {
  const float* x      = (const float*)d_in[0];
  const int*   src    = (const int*)d_in[1];
  const int*   dst    = (const int*)d_in[2];
  const int*   et     = (const int*)d_in[3];
  const float* norm   = (const float*)d_in[4];
  const float* weight = (const float*)d_in[5];
  const float* wcomp  = (const float*)d_in[6];
  const float* bias   = (const float*)d_in[7];
  const float* loopw  = (const float*)d_in[8];
  float* out = (float*)d_out;

  const int n_nodes = in_sizes[0] / DFEAT;
  const int n_edges = in_sizes[1];

  // Workspace layout: [wpack bf16 2304x256][agg f32, up to 8 x N x 256]
  unsigned short* wpack = (unsigned short*)d_ws;
  size_t packBytes = ((size_t)KTOTAL * DFEAT * 2 + 255) & ~(size_t)255;
  float* agg = (float*)((char*)d_ws + packBytes);
  size_t perBase = (size_t)n_nodes * DFEAT * sizeof(float);
  size_t avail = (ws_size > packBytes) ? ws_size - packBytes : 0;
  int nbFit = (int)(avail / perBase);
  if (nbFit > NBASES) nbFit = NBASES;
  if (nbFit < 1) nbFit = 1;   // assume harness ws is at least one basis buffer

  rgcn_pack_w<<<KTOTAL, 256, 0, stream>>>(weight, loopw, wpack);

  const int nPasses = (NBASES + nbFit - 1) / nbFit;
  const int gemmGrid = (n_nodes + 31) / 32;
  for (int p = 0; p < nPasses; ++p) {
    int b0 = p * nbFit;
    int nb = (NBASES - b0 < nbFit) ? (NBASES - b0) : nbFit;
    size_t n4 = (size_t)nb * n_nodes * (DFEAT / 4);
    rgcn_zero<<<2048, 256, 0, stream>>>((float4*)agg, n4);
    rgcn_scatter<<<n_edges, 256, 0, stream>>>(x, src, dst, et, norm, wcomp,
                                              agg, b0, nb, n_nodes, n_edges);
    int isLast = (p == nPasses - 1);
    rgcn_gemm<<<gemmGrid, 256, 0, stream>>>(agg, x, wpack, bias, out, n_nodes,
                                            b0, nb, /*include_x=*/isLast,
                                            /*first=*/p == 0, /*last=*/isLast);
  }
}
